// VQEmbeddingMovingAverage_85873576116442
// MI455X (gfx1250) — matrix operand order repaired and checked by reading, not compiled, not run on hardware
//
#include <hip/hip_runtime.h>
#include <hip/hip_bf16.h>

// VQ-VAE EMA codebook update for MI455X (gfx1250, wave32).
// Distance GEMM (32768x1024x256) done with v_wmma_f32_16x16x32_bf16 using an
// error-compensated bf16 hi/lo split (~fp32 accuracy for argmin purposes).

typedef __attribute__((ext_vector_type(16))) __bf16 v16bf;
typedef __attribute__((ext_vector_type(8)))  float  v8f;

#define KCODES 1024
#define DIM    256
#define DECAY_F 0.99f
#define ONE_MINUS_DECAY_F 0.01f

// ---------------------------------------------------------------------------
// Prep: transpose codebook to [D][K] in bf16 hi/lo, compute ||c_k||^2 in f32.
// One block (256 threads) per codebook row k.
// ---------------------------------------------------------------------------
__global__ __launch_bounds__(DIM) void vq_prep_codebook(
    const float* __restrict__ emb,     // [K, D]
    __bf16* __restrict__ cbT_hi,       // [D, K]
    __bf16* __restrict__ cbT_lo,       // [D, K]
    float* __restrict__ c_sqr)         // [K]
{
    const int k = blockIdx.x;
    const int d = threadIdx.x;
    const float v = emb[k * DIM + d];
    const __bf16 h = (__bf16)v;
    cbT_hi[d * KCODES + k] = h;
    cbT_lo[d * KCODES + k] = (__bf16)(v - (float)h);

    __shared__ float red[DIM];
    red[d] = v * v;
    __syncthreads();
    #pragma unroll
    for (int s = DIM / 2; s > 0; s >>= 1) {
        if (d < s) red[d] += red[d + s];
        __syncthreads();
    }
    if (d == 0) c_sqr[k] = red[0];
}

// ---------------------------------------------------------------------------
// Main: nearest-code search. Each wave32 owns a 16-row tile of x, keeps the
// bf16 hi/lo A tile resident in VGPRs, and sweeps all 64 column tiles of the
// transposed codebook with 3 bf16 WMMAs per 32-deep K chunk.
// Block = 128 threads = 4 waves = 64 rows. Grid = N/64 (N divisible by 64).
// ---------------------------------------------------------------------------
__global__ __launch_bounds__(128) void vq_argmin_wmma(
    const float*  __restrict__ x,       // [N, D]
    const __bf16* __restrict__ cbT_hi,  // [D, K]
    const __bf16* __restrict__ cbT_lo,  // [D, K]
    const float*  __restrict__ c_sqr,   // [K]
    int* __restrict__ idx_out)          // [N]
{
    const int lane = threadIdx.x & 31;
    const int wave = threadIdx.x >> 5;
    const int m0   = blockIdx.x * 64 + wave * 16;
    const int row  = m0 + (lane & 15);
    const int hsel = lane >> 4;            // 0: K {0..7,16..23}, 1: K {8..15,24..31}

    // Load and split the A tile: 8 chunks of 16x32, bf16 hi + lo (resident).
    v16bf a_hi[8], a_lo[8];
    const float* xr = x + (size_t)row * DIM;
    #pragma unroll
    for (int c = 0; c < 8; ++c) {
        const int d0 = c * 32 + hsel * 8;
        #pragma unroll
        for (int e = 0; e < 8; ++e) {
            const float v0 = xr[d0 + e];
            const float v1 = xr[d0 + 16 + e];
            const __bf16 h0 = (__bf16)v0;
            const __bf16 h1 = (__bf16)v1;
            a_hi[c][e]     = h0;
            a_hi[c][8 + e] = h1;
            a_lo[c][e]     = (__bf16)(v0 - (float)h0);
            a_lo[c][8 + e] = (__bf16)(v1 - (float)h1);
        }
    }

    float best[8];
    int   bidx[8];
    #pragma unroll
    for (int r = 0; r < 8; ++r) { best[r] = 3.4e38f; bidx[r] = 0; }

    for (int t = 0; t < KCODES / 16; ++t) {
        const int n0 = t * 16;
        v8f acc = {};
        #pragma unroll
        for (int c = 0; c < 8; ++c) {
            // B layout: lane = N column (lane & 15), K {0..7,16..23} / {8..15,24..31} per half.
            v16bf B_hi, B_lo;
            #pragma unroll
            for (int e = 0; e < 8; ++e) {
                const size_t boff = (size_t)(c * 32 + hsel * 8 + e) * KCODES + n0 + (lane & 15);
                B_hi[e]     = cbT_hi[boff];
                B_hi[8 + e] = cbT_hi[boff + 16 * KCODES];
                B_lo[e]     = cbT_lo[boff];
                B_lo[8 + e] = cbT_lo[boff + 16 * KCODES];
            }
            acc = __builtin_amdgcn_wmma_f32_16x16x32_bf16(
                false, a_hi[c], false, B_hi, (short)0, acc, false, false);
            acc = __builtin_amdgcn_wmma_f32_16x16x32_bf16(
                false, a_hi[c], false, B_lo, (short)0, acc, false, false);
            acc = __builtin_amdgcn_wmma_f32_16x16x32_bf16(
                false, a_lo[c], false, B_hi, (short)0, acc, false, false);
        }
        // dist ordering: c_sqr[n] - 2*dot  (||x||^2 is constant per row).
        const int   n  = n0 + (lane & 15);
        const float cs = c_sqr[n];
        #pragma unroll
        for (int r = 0; r < 8; ++r) {
            const float dist = cs - 2.0f * acc[r];
            if (dist < best[r]) { best[r] = dist; bidx[r] = n; }
        }
    }

    // Min+index reduction over N across each 16-lane half (M = r + 8*hsel).
    #pragma unroll
    for (int r = 0; r < 8; ++r) {
        float v = best[r];
        int   i = bidx[r];
        #pragma unroll
        for (int off = 1; off < 16; off <<= 1) {
            const float ov = __shfl_xor(v, off, 32);
            const int   oi = __shfl_xor(i, off, 32);
            if (ov < v || (ov == v && oi < i)) { v = ov; i = oi; }
        }
        if ((lane & 15) == 0) idx_out[m0 + hsel * 8 + r] = i;
    }
}

// ---------------------------------------------------------------------------
// Zero scratch accumulators.
// ---------------------------------------------------------------------------
__global__ void vq_zero(float* __restrict__ p, int n)
{
    const int i = blockIdx.x * blockDim.x + threadIdx.x;
    if (i < n) p[i] = 0.0f;
}

// ---------------------------------------------------------------------------
// Fused: z_q_x gather + segment-sum scatter (counts, dw) via f32 atomics.
// One block (256 threads) per row n.
// ---------------------------------------------------------------------------
__global__ __launch_bounds__(DIM) void vq_gather_scatter(
    const float* __restrict__ x,        // [N, D]
    const float* __restrict__ emb,      // [K, D]
    const int*   __restrict__ idx,      // [N]
    float* __restrict__ zq,             // [N, D]
    float* __restrict__ counts,         // [K]
    float* __restrict__ dw)             // [K, D]
{
    const int n = blockIdx.x;
    const int d = threadIdx.x;
    const int k = idx[n];
    const float xv = x[(size_t)n * DIM + d];
    zq[(size_t)n * DIM + d] = emb[k * DIM + d];
    atomicAdd(&dw[k * DIM + d], xv);
    if (d == 0) atomicAdd(&counts[k], 1.0f);
}

// ---------------------------------------------------------------------------
// EMA finalize: new_count, new_w, new_embedding. One block per k.
// ---------------------------------------------------------------------------
__global__ __launch_bounds__(DIM) void vq_finalize(
    const float* __restrict__ ema_count, // [K]
    const float* __restrict__ ema_w,     // [K, D]
    const float* __restrict__ counts,    // [K]
    const float* __restrict__ dw,        // [K, D]
    float* __restrict__ new_emb,         // [K, D]
    float* __restrict__ new_count,       // [K]
    float* __restrict__ new_w)           // [K, D]
{
    const int k = blockIdx.x;
    const int d = threadIdx.x;
    const float nc = DECAY_F * ema_count[k] + ONE_MINUS_DECAY_F * counts[k];
    const float nw = DECAY_F * ema_w[k * DIM + d] + ONE_MINUS_DECAY_F * dw[k * DIM + d];
    new_w[k * DIM + d]   = nw;
    new_emb[k * DIM + d] = nw / nc;
    if (d == 0) new_count[k] = nc;
}

// ---------------------------------------------------------------------------
// z_q_x_bar gather from updated codebook. One block per row n.
// ---------------------------------------------------------------------------
__global__ __launch_bounds__(DIM) void vq_gather_bar(
    const float* __restrict__ new_emb,  // [K, D]
    const int*   __restrict__ idx,      // [N]
    float* __restrict__ zq_bar)         // [N, D]
{
    const int n = blockIdx.x;
    const int d = threadIdx.x;
    const int k = idx[n];
    zq_bar[(size_t)n * DIM + d] = new_emb[k * DIM + d];
}

// ---------------------------------------------------------------------------
extern "C" void kernel_launch(void* const* d_in, const int* in_sizes, int n_in,
                              void* d_out, int out_size, void* d_ws, size_t ws_size,
                              hipStream_t stream)
{
    const float* z_e_x     = (const float*)d_in[0];  // [32*1024, 256]
    const float* embedding = (const float*)d_in[1];  // [1024, 256]
    const float* ema_count = (const float*)d_in[2];  // [1024]
    const float* ema_w     = (const float*)d_in[3];  // [1024, 256]

    const int NE = in_sizes[0];          // 8388608
    const int KD = in_sizes[1];          // 262144
    const int Kc = in_sizes[2];          // 1024
    const int N  = NE / DIM;             // 32768

    // Output layout (flat, return order):
    float* out_zq    = (float*)d_out;                 // [N, D]
    float* out_zqbar = out_zq + NE;                   // [N, D]
    float* out_nemb  = out_zqbar + NE;                // [K, D]
    float* out_ncnt  = out_nemb + KD;                 // [K]
    float* out_nw    = out_ncnt + Kc;                 // [K, D]

    // Workspace layout (256B-aligned blocks, ~2.2 MB total):
    char* ws = (char*)d_ws;
    int*    ws_idx   = (int*)(ws + 0);                       // 131072 B
    float*  ws_csqr  = (float*)(ws + 131072);                //   4096 B
    __bf16* ws_cbThi = (__bf16*)(ws + 135168);               // 524288 B
    __bf16* ws_cbTlo = (__bf16*)(ws + 659456);               // 524288 B
    float*  ws_cnt   = (float*)(ws + 1183744);               //   4096 B
    float*  ws_dw    = (float*)(ws + 1187840);               // 1048576 B
    // (counts + dw are contiguous: zero them together)

    vq_prep_codebook<<<KCODES, DIM, 0, stream>>>(embedding, ws_cbThi, ws_cbTlo, ws_csqr);

    const int nzero = Kc + KD;  // counts + dw
    vq_zero<<<(nzero + 255) / 256, 256, 0, stream>>>(ws_cnt, nzero);

    vq_argmin_wmma<<<N / 64, 128, 0, stream>>>(z_e_x, ws_cbThi, ws_cbTlo, ws_csqr, ws_idx);

    vq_gather_scatter<<<N, DIM, 0, stream>>>(z_e_x, embedding, ws_idx,
                                             out_zq, ws_cnt, ws_dw);

    vq_finalize<<<KCODES, DIM, 0, stream>>>(ema_count, ema_w, ws_cnt, ws_dw,
                                            out_nemb, out_ncnt, out_nw);

    vq_gather_bar<<<N, DIM, 0, stream>>>(out_nemb, ws_idx, out_zqbar);
}
